// MultiheadRelativeAttention_42889543417877
// MI455X (gfx1250) — matrix operand order repaired and checked by reading, not compiled, not run on hardware
//
#include <hip/hip_runtime.h>
#include <stdint.h>

#define L_SEQ 1024
#define BATCH 4
#define EMB   1024
#define HEADS 16
#define DHEAD 64
#define BHN   (BATCH*HEADS)    /* 64  */
#define M_ROWS (L_SEQ*BATCH)   /* 4096 */
#define N3    (3*EMB)          /* 3072 */

typedef uint32_t u32;
typedef uint64_t u64;
typedef __attribute__((ext_vector_type(16))) _Float16 v16h;
typedef __attribute__((ext_vector_type(8)))  _Float16 v8h;
typedef __attribute__((ext_vector_type(8)))  float    v8f;
typedef __attribute__((ext_vector_type(4)))  u32      v4u;
typedef __attribute__((ext_vector_type(8)))  int      v8i;
typedef __attribute__((ext_vector_type(4)))  int      v4i;

__device__ __forceinline__ v16h cat8(v8h lo, v8h hi) {
  return __builtin_shufflevector(lo, hi, 0,1,2,3,4,5,6,7,8,9,10,11,12,13,14,15);
}

__device__ __forceinline__ v8f wmma16(v16h a, v16h b, v8f c) {
  // D = A(16x32 f16) * B(32x16 f16) + C(16x16 f32)
  return __builtin_amdgcn_wmma_f32_16x16x32_f16(false, a, false, b, (short)0, c, false, false);
}

// A fragment (16x32 f16) from row-major A[ld]: lane holds row m0+lane%16.
// halves 0..7 = K kb..kb+7, halves 8..15 = K kb+16..kb+23, kb = k0 + 8*(lane>=16)
__device__ __forceinline__ v16h load_a(const _Float16* A, int ld, int m0, int k0, int lane) {
  int lr = lane & 15;
  int kb = k0 + ((lane >> 4) << 3);
  const _Float16* p = A + (size_t)(m0 + lr) * ld + kb;
  v8h lo = *(const v8h*)(p);
  v8h hi = *(const v8h*)(p + 16);
  return cat8(lo, hi);
}

// B fragment (32x16 f16) for C = X * W^T: B[k][n] = W[n][k]; row-major W[ld].
// lane holds col n0+lane%16; halves 0..7 = K kb..kb+7, halves 8..15 = K kb+16..kb+23, kb = k0 + 8*(lane>=16)
__device__ __forceinline__ v16h load_bt(const _Float16* W, int ld, int n0, int k0, int lane) {
  int lr = lane & 15;
  int kb = k0 + ((lane >> 4) << 3);
  const _Float16* p = W + (size_t)(n0 + lr) * ld + kb;
  v8h lo = *(const v8h*)(p);
  v8h hi = *(const v8h*)(p + 16);
  return cat8(lo, hi);
}

// ---- Tensor Data Mover: 2D tile (d0 elems/row x d1 rows, f16) global -> LDS.
// D# per ISA 8.3/8.4: group0 = {count=1 | lds_addr | global_addr | type=2},
// group1 = {data_size=2B, tensor dims == tile dims (all in-bounds), stride0 = row pitch}.
#if defined(__clang_major__) && (__clang_major__ >= 23)
#define TDM_CALL(g0, g1) __builtin_amdgcn_tensor_load_to_lds(g0, g1, (v4i)0, (v4i)0, (v8i)0, 0)
#else
#define TDM_CALL(g0, g1) __builtin_amdgcn_tensor_load_to_lds(g0, g1, (v4i)0, (v4i)0, 0)
#endif

__device__ __forceinline__ void tdm_load_2d(u32 lds_off, const _Float16* gptr,
                                            u32 d0, u32 d1, u32 stride0) {
  u64 ga = (u64)(uintptr_t)gptr;
  v4u g0 = { 1u,                                   // count=1 (valid user descriptor)
             lds_off,                              // lds_addr (bytes)
             (u32)ga,                              // global_addr[31:0]
             (u32)((ga >> 32) & 0x01FFFFFFu) | 0x80000000u };  // ga[56:32] | type=2
  v8i g1 = { (int)0x00010000,                                        // wg_mask=0, data_size=1 (2B)
             (int)((d0 & 0xffffu) << 16),                            // tensor_dim0 lo16
             (int)(((d0 >> 16) & 0xffffu) | ((d1 & 0xffffu) << 16)), // dim0 hi | tensor_dim1 lo
             (int)(((d1 >> 16) & 0xffffu) | ((d0 & 0xffffu) << 16)), // dim1 hi | tile_dim0
             (int)(d1 & 0xffffu),                                    // tile_dim1 (tile_dim2=0)
             (int)stride0,                                           // tensor_dim0_stride lo32
             0, 0 };
  TDM_CALL(g0, g1);
}

__global__ void k_cast(const float* __restrict__ in, _Float16* __restrict__ out, int n) {
  int i = blockIdx.x * blockDim.x + threadIdx.x;
  if (i < n) out[i] = (_Float16)in[i];
}

// qkv = X(4096x1024) @ W^T(1024x3072) + bias; scatter into qh (scaled), kh, vT (transposed).
// Per-wave 32x64 output tile: 8 WMMA per K-step, B fragments amortized over 2 row-tiles.
__global__ void __launch_bounds__(256) k_inproj(const _Float16* __restrict__ X,
                                                const _Float16* __restrict__ W,
                                                const float* __restrict__ bias,
                                                _Float16* __restrict__ qh,
                                                _Float16* __restrict__ kh,
                                                _Float16* __restrict__ vT) {
  const int lane = threadIdx.x & 31;
  const int wid  = blockIdx.x * (blockDim.x >> 5) + (threadIdx.x >> 5);
  const int mt = wid / (N3 / 64);
  const int nt = wid % (N3 / 64);
  const int m0 = mt * 32, n0 = nt * 64;
  v8f acc[2][4] = {};
  for (int k0 = 0; k0 < EMB; k0 += 32) {
    v16h a0 = load_a(X, EMB, m0,      k0, lane);
    v16h a1 = load_a(X, EMB, m0 + 16, k0, lane);
    #pragma unroll
    for (int t = 0; t < 4; ++t) {
      v16h b = load_bt(W, EMB, n0 + 16 * t, k0, lane);
      acc[0][t] = wmma16(a0, b, acc[0][t]);
      acc[1][t] = wmma16(a1, b, acc[1][t]);
    }
  }
  const int hf = lane >> 4, lr = lane & 15;
  #pragma unroll
  for (int t = 0; t < 4; ++t) {
    int n  = n0 + 16 * t + lr;
    int sec = n >> 10, e = n & 1023;
    int h = e >> 6, d = e & 63;
    float bv = bias[n];
    #pragma unroll
    for (int g = 0; g < 2; ++g) {
      #pragma unroll
      for (int i = 0; i < 8; ++i) {
        int m = m0 + g * 16 + i + hf * 8;   // C layout: row = vgpr + 8*half
        int l = m >> 2, b = m & 3;          // m = l*B + b
        int bh = b * HEADS + h;
        float v = acc[g][t][i] + bv;
        if (sec == 0)      qh[((size_t)bh * L_SEQ + l) * DHEAD + d] = (_Float16)(v * 0.125f);
        else if (sec == 1) kh[((size_t)bh * L_SEQ + l) * DHEAD + d] = (_Float16)v;
        else               vT[((size_t)bh * DHEAD + d) * L_SEQ + l] = (_Float16)v;
      }
    }
  }
}

// LDS map (dynamic extern shared => LDS offset 0):
//   [0,      65536)  QE  [32][1024] f16
//   [65536,  67584)  P staging, 2 waves x [16][32] f16
//   [67584,  83968)  K/V tile double buffer: per buf p: K[32][64] @ +p*8192, V[64][32] @ +4096+p*8192
#define LDS_QE   0u
#define LDS_PST  65536u
#define LDS_KV   67584u
#define KV_BUFSZ 8192u

// Fused relative attention: block = (bh, 32 rows), 2 waves x 16 rows.
// Phase A: QE[r][m] = q_r . er_m (WMMA -> LDS). Phase B: flash softmax; K/V tiles
// staged by the Tensor Data Mover, double-buffered on TENSORcnt.
__global__ void __launch_bounds__(64) k_attn(const _Float16* __restrict__ qh,
                                             const _Float16* __restrict__ kh,
                                             const _Float16* __restrict__ vT,
                                             const _Float16* __restrict__ er,
                                             _Float16* __restrict__ o16) {
  extern __shared__ _Float16 smem[];
  const int lane = threadIdx.x & 31;
  const int wid  = threadIdx.x >> 5;         // 0..1
  const int rb   = blockIdx.x;               // 0..31
  const int bh   = blockIdx.y;               // 0..63
  const int i0   = rb * 32 + wid * 16;
  const int hf   = lane >> 4, lr = lane & 15;

  _Float16* QE  = smem;
  _Float16* Pst = smem + LDS_PST / 2 + wid * (16 * 32);

  const _Float16* qP = qh + (size_t)bh * L_SEQ * DHEAD;
  const _Float16* kP = kh + (size_t)bh * L_SEQ * DHEAD;
  const _Float16* vP = vT + (size_t)bh * DHEAD * L_SEQ;

  // Kick off the TDM for the first K/V tile; it streams while we compute QE.
  if (wid == 0) {
    tdm_load_2d(LDS_KV,         kP, 64, 32, 64);     // K rows 0..31 (32x64 f16)
    tdm_load_2d(LDS_KV + 4096u, vP, 32, 64, 1024);   // vT cols 0..31 (64 rows x 32)
  }

  v16h aq0 = load_a(qP, DHEAD, i0, 0,  lane);
  v16h aq1 = load_a(qP, DHEAD, i0, 32, lane);

  // ---- Phase A: QE for this wave's 16 rows, all m in [0,1024)
  for (int mt = 0; mt < L_SEQ; mt += 16) {
    v8f acc = {};
    acc = wmma16(aq0, load_bt(er, DHEAD, mt, 0,  lane), acc);
    acc = wmma16(aq1, load_bt(er, DHEAD, mt, 32, lane), acc);
    #pragma unroll
    for (int i = 0; i < 8; ++i) {
      int r = wid * 16 + i + hf * 8;
      QE[(size_t)r * L_SEQ + mt + lr] = (_Float16)acc[i];
    }
  }
  __builtin_amdgcn_wave_barrier();

  v8f m_run, l_run;
  #pragma unroll
  for (int i = 0; i < 8; ++i) { m_run[i] = -3.0e38f; l_run[i] = 0.0f; }
  v8f o0 = {}, o1 = {}, o2 = {}, o3 = {};

  // ---- Phase B: stream over 32-wide key/value column tiles
  for (int j0 = 0; j0 < L_SEQ; j0 += 32) {
    const int p = (j0 >> 5) & 1;
    if (wid == 0) __builtin_amdgcn_s_wait_tensorcnt(0);  // buf p landed
    __syncthreads();                                     // visible to both waves;
                                                         // prior reads of buf p^1 done
    if (wid == 0 && (j0 + 32) < L_SEQ) {                 // prefetch next tile into p^1
      const u32 nb = LDS_KV + (u32)(p ^ 1) * KV_BUFSZ;
      tdm_load_2d(nb,         kP + (size_t)(j0 + 32) * DHEAD, 64, 32, 64);
      tdm_load_2d(nb + 4096u, vP + (j0 + 32),                 32, 64, 1024);
    }
    const _Float16* Kt = smem + (LDS_KV + p * KV_BUFSZ) / 2;
    const _Float16* Vt = Kt + 4096 / 2;

    v8f s0 = {}, s1 = {};
    s0 = wmma16(aq0, load_bt(Kt, DHEAD, 0,  0,  lane), s0);
    s0 = wmma16(aq1, load_bt(Kt, DHEAD, 0,  32, lane), s0);
    s1 = wmma16(aq0, load_bt(Kt, DHEAD, 16, 0,  lane), s1);
    s1 = wmma16(aq1, load_bt(Kt, DHEAD, 16, 32, lane), s1);

    // srel[i,j] = (j<=i) ? QE[i, 1023 + j - i] : 0
    #pragma unroll
    for (int i = 0; i < 8; ++i) {
      int r   = i + hf * 8;
      int gi  = i0 + r;
      int row = wid * 16 + r;
      int ma  = (L_SEQ - 1) + (j0 + lr) - gi;
      int mb  = ma + 16;
      if (ma < L_SEQ) s0[i] += (float)QE[(size_t)row * L_SEQ + ma];
      if (mb < L_SEQ) s1[i] += (float)QE[(size_t)row * L_SEQ + mb];
    }

    // online softmax: row stats via 16-lane xor reductions (rows live per-half)
    v8f alpha, p0, p1;
    #pragma unroll
    for (int i = 0; i < 8; ++i) {
      float t = fmaxf(s0[i], s1[i]);
      #pragma unroll
      for (int msk = 1; msk < 16; msk <<= 1) t = fmaxf(t, __shfl_xor(t, msk, 32));
      float mn = fmaxf(m_run[i], t);
      float al = __expf(m_run[i] - mn);
      float e0 = __expf(s0[i] - mn);
      float e1 = __expf(s1[i] - mn);
      float rs = e0 + e1;
      #pragma unroll
      for (int msk = 1; msk < 16; msk <<= 1) rs += __shfl_xor(rs, msk, 32);
      l_run[i] = l_run[i] * al + rs;
      m_run[i] = mn;
      alpha[i] = al; p0[i] = e0; p1[i] = e1;
    }
    #pragma unroll
    for (int i = 0; i < 8; ++i) {
      o0[i] *= alpha[i]; o1[i] *= alpha[i]; o2[i] *= alpha[i]; o3[i] *= alpha[i];
    }

    // transpose P: C-fragment -> row-major LDS -> A-fragment
    #pragma unroll
    for (int i = 0; i < 8; ++i) {
      int r = i + hf * 8;
      Pst[r * 32 + lr]      = (_Float16)p0[i];
      Pst[r * 32 + 16 + lr] = (_Float16)p1[i];
    }
    __builtin_amdgcn_wave_barrier();
    v16h pa;
    {
      const _Float16* pp = Pst + lr * 32 + (hf << 3);
      v8h lo = *(const v8h*)(pp);
      v8h hi = *(const v8h*)(pp + 16);
      pa = cat8(lo, hi);
    }
    __builtin_amdgcn_wave_barrier();

    // O += P @ V  (V tile already transposed in LDS: Vt[d][jloc])
    o0 = wmma16(pa, load_bt(Vt, 32,  0, 0, lane), o0);
    o1 = wmma16(pa, load_bt(Vt, 32, 16, 0, lane), o1);
    o2 = wmma16(pa, load_bt(Vt, 32, 32, 0, lane), o2);
    o3 = wmma16(pa, load_bt(Vt, 32, 48, 0, lane), o3);
  }

  // normalize + store to o16[(l*B + b)*E + h*64 + d]
  const int b = bh >> 4, h = bh & 15;
  #pragma unroll
  for (int i = 0; i < 8; ++i) {
    int gi = i0 + i + hf * 8;
    float inv = 1.0f / l_run[i];
    size_t base = ((size_t)gi * BATCH + b) * EMB + h * DHEAD + lr;
    o16[base +  0] = (_Float16)(o0[i] * inv);
    o16[base + 16] = (_Float16)(o1[i] * inv);
    o16[base + 32] = (_Float16)(o2[i] * inv);
    o16[base + 48] = (_Float16)(o3[i] * inv);
  }
}

// out = X(4096x1024) @ Wo^T(1024x1024) + bias -> f32, per-wave 32x64 tile
__global__ void __launch_bounds__(256) k_outproj(const _Float16* __restrict__ X,
                                                 const _Float16* __restrict__ W,
                                                 const float* __restrict__ bias,
                                                 float* __restrict__ out) {
  const int lane = threadIdx.x & 31;
  const int wid  = blockIdx.x * (blockDim.x >> 5) + (threadIdx.x >> 5);
  const int mt = wid / (EMB / 64);
  const int nt = wid % (EMB / 64);
  const int m0 = mt * 32, n0 = nt * 64;
  v8f acc[2][4] = {};
  for (int k0 = 0; k0 < EMB; k0 += 32) {
    v16h a0 = load_a(X, EMB, m0,      k0, lane);
    v16h a1 = load_a(X, EMB, m0 + 16, k0, lane);
    #pragma unroll
    for (int t = 0; t < 4; ++t) {
      v16h b = load_bt(W, EMB, n0 + 16 * t, k0, lane);
      acc[0][t] = wmma16(a0, b, acc[0][t]);
      acc[1][t] = wmma16(a1, b, acc[1][t]);
    }
  }
  const int hf = lane >> 4, lr = lane & 15;
  #pragma unroll
  for (int t = 0; t < 4; ++t) {
    int n = n0 + 16 * t + lr;
    float bv = bias[n];
    #pragma unroll
    for (int g = 0; g < 2; ++g) {
      #pragma unroll
      for (int i = 0; i < 8; ++i) {
        int m = m0 + g * 16 + i + hf * 8;
        out[(size_t)m * EMB + n] = acc[g][t][i] + bv;
      }
    }
  }
}

extern "C" void kernel_launch(void* const* d_in, const int* in_sizes, int n_in,
                              void* d_out, int out_size, void* d_ws, size_t ws_size,
                              hipStream_t stream) {
  const float* query  = (const float*)d_in[0];
  const float* relpos = (const float*)d_in[1];
  const float* wi     = (const float*)d_in[2];
  const float* bi     = (const float*)d_in[3];
  const float* wo     = (const float*)d_in[4];
  const float* bo     = (const float*)d_in[5];

  char* ws = (char*)d_ws;
  size_t off = 0;
  auto alloc = [&](size_t bytes) -> char* {
    char* p = ws + off;
    off += (bytes + 255) & ~(size_t)255;
    return p;
  };
  _Float16* q16  = (_Float16*)alloc((size_t)M_ROWS * EMB * 2);        // 8 MB
  _Float16* w16  = (_Float16*)alloc((size_t)N3 * EMB * 2);            // 6 MB
  _Float16* wo16 = (_Float16*)alloc((size_t)EMB * EMB * 2);           // 2 MB
  _Float16* er16 = (_Float16*)alloc((size_t)L_SEQ * DHEAD * 2);       // 128 KB
  _Float16* qhh  = (_Float16*)alloc((size_t)BHN * L_SEQ * DHEAD * 2); // 8 MB
  _Float16* khh  = (_Float16*)alloc((size_t)BHN * L_SEQ * DHEAD * 2); // 8 MB
  _Float16* vTT  = (_Float16*)alloc((size_t)BHN * DHEAD * L_SEQ * 2); // 8 MB
  _Float16* o16  = q16;  // q16 dead after in_proj; reuse for attention output

  k_cast<<<(M_ROWS * EMB + 255) / 256, 256, 0, stream>>>(query, q16, M_ROWS * EMB);
  k_cast<<<(N3 * EMB + 255) / 256, 256, 0, stream>>>(wi, w16, N3 * EMB);
  k_cast<<<(EMB * EMB + 255) / 256, 256, 0, stream>>>(wo, wo16, EMB * EMB);
  k_cast<<<(L_SEQ * DHEAD + 255) / 256, 256, 0, stream>>>(relpos, er16, L_SEQ * DHEAD);

  // 128 m32-tiles x 48 n64-tiles = 6144 waves / 8 per block
  k_inproj<<<768, 256, 0, stream>>>(q16, w16, bi, qhh, khh, vTT);

  dim3 gattn(32, 64);  // row-blocks x bh
  size_t lds = (size_t)LDS_KV + 2 * KV_BUFSZ;  // QE + P staging + TDM double buffer = 82 KB
  k_attn<<<gattn, 64, lds, stream>>>(qhh, khh, vTT, er16, o16);

  // 128 m32-tiles x 16 n64-tiles = 2048 waves / 8 per block
  k_outproj<<<256, 256, 0, stream>>>(o16, wo16, bo, (float*)d_out);
}
